// InteractionLayer_54563264528557
// MI455X (gfx1250) — compile-verified, matrix operand order checked by reading, not run on hardware
//
#include <hip/hip_runtime.h>
#include <math.h>

// ---------------------------------------------------------------------------
// Shapes (from reference): N=10000, E=160000, F=64, B=8, S=9, A=16, DQ=16
// ---------------------------------------------------------------------------
static constexpr int FDIM  = 64;
static constexpr int SDIM  = 9;
static constexpr int BDIM  = 8;
static constexpr int ADIM  = 16;
static constexpr int DQDIM = 16;
static constexpr int KAGG  = FDIM * SDIM;   // 576
static constexpr int NLAYERS = 2;
static constexpr float RMAX = 5.0f;

typedef __attribute__((ext_vector_type(16))) _Float16 v16h;
typedef __attribute__((ext_vector_type(8)))  float    v8f;

__device__ __forceinline__ _Float16 f2h(float f) { return (_Float16)f; }
__device__ __forceinline__ float silu_f(float v) { return v / (1.0f + expf(-v)); }

__device__ __forceinline__ float atomicMaxFloat(float* addr, float val) {
    unsigned int* ua = (unsigned int*)addr;
    unsigned int old = *ua;
    while (__uint_as_float(old) < val) {
        unsigned int assumed = old;
        old = atomicCAS(ua, assumed, __float_as_uint(val));
        if (old == assumed) break;
    }
    return __uint_as_float(old);
}

// ---------------------------------------------------------------------------
// Edge geometry: radial Bessel * polynomial cutoff, real spherical harmonics.
// ---------------------------------------------------------------------------
__global__ void edge_geom_kernel(const float* __restrict__ vec,
                                 const float* __restrict__ freqs,
                                 float* __restrict__ es,   // E x 8
                                 float* __restrict__ Yb,   // E x 9
                                 int E_) {
    int e = blockIdx.x * blockDim.x + threadIdx.x;
    if (e >= E_) return;
    float vx = vec[e*3+0], vy = vec[e*3+1], vz = vec[e*3+2];
    float r  = sqrtf(vx*vx + vy*vy + vz*vz);
    float ir = 1.0f / r;
    float x = vx*ir, y = vy*ir, z = vz*ir;
    float t = r / RMAX;
    float t2 = t*t, t3 = t2*t, t6 = t3*t3, t7 = t6*t, t8 = t7*t;
    float fc = (t < 1.0f) ? (1.0f - 28.0f*t6 + 48.0f*t7 - 21.0f*t8) : 0.0f;
    float scale = sqrtf(2.0f / RMAX) * ir * fc;
#pragma unroll
    for (int b = 0; b < BDIM; ++b)
        es[e*BDIM + b] = scale * sinf(r * freqs[b]);
    const float s3 = 1.7320508075688772f, s15 = 3.8729833462074170f, s5 = 2.2360679774997896f;
    float* Ye = Yb + (size_t)e * SDIM;
    Ye[0] = 1.0f;
    Ye[1] = s3 * x;  Ye[2] = s3 * y;  Ye[3] = s3 * z;
    Ye[4] = s15 * x * y;  Ye[5] = s15 * y * z;
    Ye[6] = 0.5f * s5 * (3.0f*z*z - 1.0f);
    Ye[7] = s15 * x * z;  Ye[8] = 0.5f * s15 * (x*x - y*y);
}

// ---------------------------------------------------------------------------
// Per-layer init / zero
// ---------------------------------------------------------------------------
__global__ void init_nodes_kernel(float* __restrict__ mbuf, float* __restrict__ denom, int Nn) {
    int i = blockIdx.x * blockDim.x + threadIdx.x;
    if (i < Nn) { mbuf[i] = -3.0e38f; denom[i] = 0.0f; }
}

__global__ void zero_kernel(float* __restrict__ p, long long n) {
    long long i = (long long)blockIdx.x * blockDim.x + threadIdx.x;
    long long stride = (long long)gridDim.x * blockDim.x;
    for (; i < n; i += stride) p[i] = 0.0f;
}

// ---------------------------------------------------------------------------
// q = x @ Wq   (N x 16)
// ---------------------------------------------------------------------------
__global__ void q_kernel(const float* __restrict__ x, const float* __restrict__ Wq,
                         float* __restrict__ q, int Nn) {
    int i = blockIdx.x * blockDim.x + threadIdx.x;
    if (i >= Nn * DQDIM) return;
    int n = i >> 4, d = i & 15;
    float acc = 0.0f;
#pragma unroll 8
    for (int f = 0; f < FDIM; ++f) acc += x[n*FDIM + f] * Wq[f*DQDIM + d];
    q[i] = acc;
}

// ---------------------------------------------------------------------------
// Per-edge: radial MLP -> filter; logit = <q[dst], es@Wk>/sqrt(DQ);
// segment-max over dst via float atomic max.
// ---------------------------------------------------------------------------
__global__ void edge_mlp_kernel(const float* __restrict__ es, const float* __restrict__ Yb,
                                const float* __restrict__ W1, const float* __restrict__ b1,
                                const float* __restrict__ W2, const float* __restrict__ b2,
                                const float* __restrict__ W3, const float* __restrict__ b3,
                                const float* __restrict__ Wk,
                                const float* __restrict__ q, const int* __restrict__ dst,
                                float* __restrict__ filt, float* __restrict__ logit,
                                float* __restrict__ mbuf, int E_) {
    int e = blockIdx.x * blockDim.x + threadIdx.x;
    if (e >= E_) return;
    float s[BDIM];
#pragma unroll
    for (int b = 0; b < BDIM; ++b) s[b] = es[e*BDIM + b];
    float h1[8];
#pragma unroll
    for (int j = 0; j < 8; ++j) {
        float acc = b1[j];
#pragma unroll
        for (int b = 0; b < BDIM; ++b) acc += s[b] * W1[b*8 + j];
        h1[j] = silu_f(acc);
    }
    float h2[8];
#pragma unroll
    for (int j = 0; j < 8; ++j) {
        float acc = b2[j];
#pragma unroll
        for (int b = 0; b < 8; ++b) acc += h1[b] * W2[b*8 + j];
        h2[j] = silu_f(acc);
    }
#pragma unroll
    for (int j = 0; j < SDIM; ++j) {
        float acc = b3[j];
#pragma unroll
        for (int b = 0; b < 8; ++b) acc += h2[b] * W3[b*SDIM + j];
        filt[(size_t)e*SDIM + j] = Yb[(size_t)e*SDIM + j] * acc;
    }
    int d = dst[e];
    float lg = 0.0f;
#pragma unroll
    for (int j = 0; j < DQDIM; ++j) {
        float acc = 0.0f;
#pragma unroll
        for (int b = 0; b < BDIM; ++b) acc += s[b] * Wk[b*DQDIM + j];
        lg += acc * q[d*DQDIM + j];
    }
    lg *= 0.25f;  // 1/sqrt(16)
    logit[e] = lg;
    atomicMaxFloat(mbuf + d, lg);
}

// ---------------------------------------------------------------------------
// a = exp(logit - m[dst]); denom[dst] += a
// ---------------------------------------------------------------------------
__global__ void edge_exp_kernel(const float* __restrict__ logit, const float* __restrict__ mbuf,
                                const int* __restrict__ dst,
                                float* __restrict__ aexp, float* __restrict__ denom, int E_) {
    int e = blockIdx.x * blockDim.x + threadIdx.x;
    if (e >= E_) return;
    int d = dst[e];
    float a = expf(logit[e] - mbuf[d]);
    aexp[e] = a;
    atomicAdd(denom + d, a);
}

// ---------------------------------------------------------------------------
// Scatter messages: agg[dst, f, s] += x[src, f] * filt[e, s] * alpha[e]
// ---------------------------------------------------------------------------
__global__ void edge_scatter_kernel(const float* __restrict__ x, const float* __restrict__ filt,
                                    const float* __restrict__ aexp, const float* __restrict__ denom,
                                    const int* __restrict__ src, const int* __restrict__ dst,
                                    float* __restrict__ agg, int E_) {
    long long i = (long long)blockIdx.x * blockDim.x + threadIdx.x;
    if (i >= (long long)E_ * FDIM) return;
    int e = (int)(i >> 6), f = (int)(i & 63);
    int sn = src[e], d = dst[e];
    float alpha = aexp[e] / (denom[d] + 1e-9f);
    float xv = x[(size_t)sn*FDIM + f] * alpha;
    const float* fe = filt + (size_t)e * SDIM;
    float* ag = agg + (size_t)d * KAGG + (size_t)f * SDIM;
#pragma unroll
    for (int s = 0; s < SDIM; ++s) atomicAdd(ag + s, xv * fe[s]);
}

// ---------------------------------------------------------------------------
// WMMA output GEMM: xout(N x 64) = agg@Wout + x@Wself + attr@Wattr
//
// One wave32 computes a 16x64 output stripe (4 C tiles) so each A fragment
// feeds 4 WMMAs. B fragments for all 4 n-tiles are staged once per block into
// LDS, pre-swizzled into the exact per-lane v16h layout (two ds_load_b128 to
// consume). All guards are wave-uniform or branch-free (row clamp); EXEC stays
// all-ones around every v_wmma as the ISA requires.
//
// A-layout (16x32 f16): lane l holds row (l&15); K elems [kb..kb+7] in halves
// 0..7 and [kb+16..kb+23] in halves 8..15, kb=(l>>4)*8.
// C/D layout: VGPR v, lane l -> row = v + 8*(l>>4), col = l&15.
// ---------------------------------------------------------------------------
__device__ __forceinline__ void cvt8(const float4 p0, const float4 p1, v16h& a, const int base) {
    a[base+0] = f2h(p0.x); a[base+1] = f2h(p0.y); a[base+2] = f2h(p0.z); a[base+3] = f2h(p0.w);
    a[base+4] = f2h(p1.x); a[base+5] = f2h(p1.y); a[base+6] = f2h(p1.z); a[base+7] = f2h(p1.w);
}

template <int K, int LDA>
__device__ __forceinline__ void gemm_seg(const float* __restrict__ A,
                                         const float* __restrict__ Bw,   // K x 64, ldb=64
                                         _Float16* __restrict__ sB,      // [4][32][16]
                                         int rowc, int kb, int lane, v8f c[4]) {
    for (int k0 = 0; k0 < K; k0 += 32) {
        __syncthreads();
        // Stage B chunk for all 4 n-tiles, pre-swizzled per lane.
#pragma unroll
        for (int it = 0; it < 8; ++it) {
            int t    = (int)threadIdx.x + it * 256;     // 0..2047
            int nt   = t >> 9;
            int rem  = t & 511;
            int ln   = rem >> 4;
            int j    = rem & 15;
            int lkb  = (ln >> 4) * 8;
            int kk   = k0 + lkb + ((j < 8) ? j : (8 + j));   // j>=8 -> kb+16+(j-8)
            int col  = nt * 16 + (ln & 15);
            float v  = (kk < K) ? Bw[(size_t)kk * FDIM + col] : 0.0f;
            sB[t] = f2h(v);
        }
        __syncthreads();

        // Load A fragment: 2x8 contiguous floats -> 4x float4 (b128 loads).
        v16h a;
        const float* Arow = A + (size_t)rowc * LDA + k0 + kb;
        const bool lo_ok = (k0 + 16 <= K);   // wave-uniform, foldable
        const bool hi_ok = (k0 + 32 <= K);
        if (lo_ok) {
            float4 p0 = *reinterpret_cast<const float4*>(Arow);
            float4 p1 = *reinterpret_cast<const float4*>(Arow + 4);
            cvt8(p0, p1, a, 0);
        } else {
#pragma unroll
            for (int j = 0; j < 8; ++j) a[j] = f2h(0.0f);
        }
        if (hi_ok) {
            float4 p2 = *reinterpret_cast<const float4*>(Arow + 16);
            float4 p3 = *reinterpret_cast<const float4*>(Arow + 20);
            cvt8(p2, p3, a, 8);
        } else {
#pragma unroll
            for (int j = 8; j < 16; ++j) a[j] = f2h(0.0f);
        }

        // 4 WMMAs per A fragment (reuse across the 16x64 stripe).
#pragma unroll
        for (int nt = 0; nt < 4; ++nt) {
            const v16h b = *reinterpret_cast<const v16h*>(&sB[(nt << 9) + (lane << 4)]);
            c[nt] = __builtin_amdgcn_wmma_f32_16x16x32_f16(
                        false, a, false, b, (short)0, c[nt], false, false);
        }
    }
}

__global__ void out_gemm_kernel(const float* __restrict__ agg,  const float* __restrict__ x,
                                const float* __restrict__ attr,
                                const float* __restrict__ Wout, const float* __restrict__ Wself,
                                const float* __restrict__ Wattr,
                                float* __restrict__ xout, int Nrows) {
    __shared__ _Float16 sB[4 * 32 * 16];   // 4 KB

    const int lane = threadIdx.x & 31;
    const int wid  = threadIdx.x >> 5;                   // 8 waves / block
    const int Mt   = (Nrows + 15) >> 4;
    int mt = blockIdx.x * 8 + wid;
    if (mt >= Mt) mt = Mt - 1;   // keep all waves alive for barriers (dup stores benign)

    const int l15  = lane & 15;
    const int half = lane >> 4;
    const int kb   = half * 8;
    const int row  = mt * 16 + l15;
    const int rowc = (row < Nrows) ? row : (Nrows - 1);  // branch-free clamp for loads

    v8f c[4] = {};
    gemm_seg<KAGG, KAGG>(agg,  Wout,  sB, rowc, kb, lane, c);   // K=576
    gemm_seg<FDIM, FDIM>(x,    Wself, sB, rowc, kb, lane, c);   // K=64
    gemm_seg<ADIM, ADIM>(attr, Wattr, sB, rowc, kb, lane, c);   // K=16 (pad->32)

#pragma unroll
    for (int nt = 0; nt < 4; ++nt) {
        const int col = nt * 16 + l15;
#pragma unroll
        for (int v = 0; v < 8; ++v) {
            int r = mt * 16 + v + half * 8;
            if (r < Nrows) xout[(size_t)r * FDIM + col] = c[nt][v];
        }
    }
}

// ---------------------------------------------------------------------------
// Host driver
// ---------------------------------------------------------------------------
extern "C" void kernel_launch(void* const* d_in, const int* in_sizes, int n_in,
                              void* d_out, int out_size, void* d_ws, size_t ws_size,
                              hipStream_t stream) {
    (void)n_in; (void)out_size; (void)ws_size;
    const float* node_feature = (const float*)d_in[0];
    const float* node_attr    = (const float*)d_in[1];
    const int*   edge_src     = (const int*)d_in[2];
    const int*   edge_dst     = (const int*)d_in[3];
    const float* edge_vec     = (const float*)d_in[4];
    const float* bessel       = (const float*)d_in[5];

    const int Nn = in_sizes[0] / FDIM;
    const int Ee = in_sizes[2];

    // Carve workspace (floats); every block is a multiple of 16 floats (64B).
    float* ws    = (float*)d_ws;
    float* es    = ws; ws += (size_t)Ee * BDIM;   // E x 8
    float* Yb    = ws; ws += (size_t)Ee * SDIM;   // E x 9
    float* filt  = ws; ws += (size_t)Ee * SDIM;   // E x 9
    float* logit = ws; ws += (size_t)Ee;
    float* aexp  = ws; ws += (size_t)Ee;
    float* mbuf  = ws; ws += (size_t)((Nn + 15) & ~15);
    float* denom = ws; ws += (size_t)((Nn + 15) & ~15);
    float* qb    = ws; ws += (size_t)Nn * DQDIM;
    float* agg   = ws; ws += (size_t)Nn * KAGG;   // N x 576
    float* xtmp  = ws; ws += (size_t)Nn * FDIM;   // layer ping buffer

    const int TB = 256;
    edge_geom_kernel<<<(Ee + TB - 1) / TB, TB, 0, stream>>>(edge_vec, bessel, es, Yb, Ee);

    const float* xin = node_feature;
    for (int L = 0; L < NLAYERS; ++L) {
        const int pb = 6 + L * 11;
        const float* W1    = (const float*)d_in[pb + 0];
        const float* b1    = (const float*)d_in[pb + 1];
        const float* W2    = (const float*)d_in[pb + 2];
        const float* b2    = (const float*)d_in[pb + 3];
        const float* W3    = (const float*)d_in[pb + 4];
        const float* b3    = (const float*)d_in[pb + 5];
        const float* Wq    = (const float*)d_in[pb + 6];
        const float* Wk    = (const float*)d_in[pb + 7];
        const float* Wout  = (const float*)d_in[pb + 8];
        const float* Wself = (const float*)d_in[pb + 9];
        const float* Wattr = (const float*)d_in[pb + 10];
        float* xout = (L == NLAYERS - 1) ? (float*)d_out : xtmp;

        init_nodes_kernel<<<(Nn + TB - 1) / TB, TB, 0, stream>>>(mbuf, denom, Nn);
        zero_kernel<<<2048, TB, 0, stream>>>(agg, (long long)Nn * KAGG);
        q_kernel<<<(Nn * DQDIM + TB - 1) / TB, TB, 0, stream>>>(xin, Wq, qb, Nn);
        edge_mlp_kernel<<<(Ee + TB - 1) / TB, TB, 0, stream>>>(
            es, Yb, W1, b1, W2, b2, W3, b3, Wk, qb, edge_dst, filt, logit, mbuf, Ee);
        edge_exp_kernel<<<(Ee + TB - 1) / TB, TB, 0, stream>>>(
            logit, mbuf, edge_dst, aexp, denom, Ee);
        long long tot = (long long)Ee * FDIM;
        edge_scatter_kernel<<<(int)((tot + TB - 1) / TB), TB, 0, stream>>>(
            xin, filt, aexp, denom, edge_src, edge_dst, agg, Ee);

        const int Mt = (Nn + 15) >> 4;
        dim3 grid((Mt + 7) / 8, 1);
        out_gemm_kernel<<<grid, TB, 0, stream>>>(agg, xin, node_attr,
                                                 Wout, Wself, Wattr, xout, Nn);
        xin = xout;
    }
}
